// VectorQuantizer_37769942401476
// MI455X (gfx1250) — matrix operand order repaired and checked by reading, not compiled, not run on hardware
//
#include <hip/hip_runtime.h>
#include <cstdint>

typedef __attribute__((ext_vector_type(16))) _Float16 v16h;
typedef __attribute__((ext_vector_type(8)))  _Float16 v8h;
typedef __attribute__((ext_vector_type(8)))  float    v8f;

#define VQ_K      1024
#define VQ_D      256
#define VQ_N      32768            // B*T = 16*2048
#define VQ_INVN   (1.0f / 32768.0f)
#define CHUNK     64               // codes staged per LDS chunk
#define NCHUNK    (VQ_K / CHUNK)   // 16
#define BPAD      264              // padded row stride in halves (528B) -> lane n hits banks 4n..4n+3

// ---------------------------------------------------------------------------
// Kernel 1: codebook fp32 -> f16, cnorm[k] = ||c_k||^2, zero counts/sse.
// ---------------------------------------------------------------------------
__global__ __launch_bounds__(256)
void vq_prep(const float* __restrict__ cb, _Float16* __restrict__ cbf16,
             float* __restrict__ cnorm, float* __restrict__ counts,
             float* __restrict__ sse)
{
    __shared__ float red[256];
    const int k = blockIdx.x;
    const int e = threadIdx.x;
    const float c = cb[k * VQ_D + e];
    cbf16[k * VQ_D + e] = (_Float16)c;
    red[e] = c * c;
    __syncthreads();
    for (int s = 128; s > 0; s >>= 1) {
        if (e < s) red[e] += red[e + s];
        __syncthreads();
    }
    if (e == 0) {
        cnorm[k] = red[0];
        counts[k] = 0.0f;
        if (k == 0) *sse = 0.0f;
    }
}

// ---------------------------------------------------------------------------
// Async-copy one 64-code chunk (64 x 512B rows) into a padded LDS buffer.
// 256 threads: 4 threads/row, 128B each, as 8 async b128 transfers.
// INST_OFFSET applies to both LDS and global addresses (ISA 08, §4.4).
// ---------------------------------------------------------------------------
__device__ __forceinline__
void async_copy_chunk(const _Float16* __restrict__ cbf16, _Float16* ldsbuf,
                      int chunk, int tid)
{
    const int row  = tid >> 2;                       // 0..63
    const int part = tid & 3;                        // 0..3 (128B slices)
    const uint64_t ga = (uint64_t)(uintptr_t)
        ((const char*)(cbf16 + ((size_t)chunk * CHUNK + row) * VQ_D) + part * 128);
    const uint32_t la = (uint32_t)(uintptr_t)
        ((char*)(ldsbuf + (size_t)row * BPAD) + part * 128);
#pragma unroll
    for (int i = 0; i < 8; ++i) {
        asm volatile("global_load_async_to_lds_b128 %0, %1, off offset:%2"
                     :: "v"(la), "v"(ga), "i"(i * 16) : "memory");
    }
}

__device__ __forceinline__ void wait_async_and_sync()
{
    asm volatile("s_wait_asynccnt 0x0" ::: "memory");
    __syncthreads();
}

// ---------------------------------------------------------------------------
// Kernel 2: main VQ. Block = 128 rows x all 1024 codes; 8 waves own 16 rows
// each. Codebook double-buffered through LDS via async-to-LDS DMA.
// Distance GEMM via v_wmma_f32_16x16x32_f16 with a register-level pipeline
// so ds_loads for fragment u+1 overlap the WMMA on fragment u.
// ---------------------------------------------------------------------------
__global__ __launch_bounds__(256)
void vq_main(const float* __restrict__ X, const float* __restrict__ cbF32,
             const _Float16* __restrict__ cbf16, const float* __restrict__ cnorm,
             float* __restrict__ outQ, float* __restrict__ outIdx,
             float* __restrict__ counts, float* __restrict__ sse)
{
    __shared__ __align__(16) _Float16 bbuf[2][CHUNK * BPAD];   // 2 x 33KB
    __shared__ float ldsD[8][16][16];
    __shared__ int   ldsI[8][16][16];
    __shared__ int   ldsWin[128];
    __shared__ float ldsS[256];

    const int tid  = threadIdx.x;
    const int wave = tid >> 5;          // 0..7 -> owns rows wave*16..wave*16+15
    const int lane = tid & 31;
    const int l16  = lane & 15;
    const int hi   = lane >> 4;
    const int rowBase = blockIdx.x * 128;
    const int waveRow = rowBase + wave * 16;

    // Prefetch chunk 0 while A fragments load.
    async_copy_chunk(cbf16, &bbuf[0][0], 0, tid);

    // ---- A fragments: 16 rows x 256 K, f16, in registers ------------------
    // ISA 16-bit A layout: lanes 0-15: row M=lane, K 0-7 (V0..3), K 16-23
    // (V4..7); lanes 16-31: K 8-15 and K 24-31.
    v16h a[8];
    {
        const float* xr = X + (size_t)(waveRow + l16) * VQ_D;
        const int off0 = hi ? 8 : 0;
#pragma unroll
        for (int s = 0; s < 8; ++s) {
            const float* p0 = xr + s * 32 + off0;
            const float* p1 = p0 + 16;
#pragma unroll
            for (int e = 0; e < 8; ++e) {
                a[s][e]     = (_Float16)p0[e];
                a[s][e + 8] = (_Float16)p1[e];
            }
        }
    }

    float bestd[8];
    int   besti[8];
#pragma unroll
    for (int j = 0; j < 8; ++j) { bestd[j] = 3.4e38f; besti[j] = 0; }

    wait_async_and_sync();              // chunk 0 resident

#pragma unroll 1
    for (int c = 0; c < NCHUNK; ++c) {
        const int buf = c & 1;
        if (c + 1 < NCHUNK)             // prefetch next chunk into other buffer
            async_copy_chunk(cbf16, &bbuf[buf ^ 1][0], c + 1, tid);

        // cnorm for the 4 tiles of this chunk: clause-grouped, complete
        // under the first tile's WMMAs.
        float cn[4];
#pragma unroll
        for (int t = 0; t < 4; ++t)
            cn[t] = cnorm[c * CHUNK + t * 16 + l16];

        // B layout: lanes 0-15 hold K 0-7,16-23; lanes 16-31 K 8-15,24-31.
        // Single base + immediate ds offsets (max 3*8448+480 < 64K).
        const _Float16* bpBase =
            &bbuf[buf][0] + (size_t)l16 * BPAD + (hi ? 8 : 0);
        auto loadB = [&](int u) -> v16h {
            const int t = u >> 3, s = u & 7;
            const v8h* b2 = (const v8h*)(bpBase + (size_t)t * 16 * BPAD + s * 32);
            v8h b0 = b2[0], b1 = b2[2];
            return __builtin_shufflevector(b0, b1,
                0, 1, 2, 3, 4, 5, 6, 7, 8, 9, 10, 11, 12, 13, 14, 15);
        };

        // Flat pipeline over the 32 fragments (4 tiles x 8 k-steps): the
        // ds_loads for u+1 issue before the WMMA on u, and the next tile's
        // loads fly during the argmin VALU burst (co-executes with XDL).
        v16h bcur = loadB(0);
        v8f acc = {};
#pragma unroll
        for (int u = 0; u < 32; ++u) {
            v16h bnext = bcur;
            if (u + 1 < 32) bnext = loadB(u + 1);
            const int s = u & 7;
            acc = __builtin_amdgcn_wmma_f32_16x16x32_f16(
                false, a[s], false, bcur, (short)0, acc, false, false);
            bcur = bnext;
            if (s == 7) {               // tile t = u>>3 finished
                const int t = u >> 3;
                const int code = c * CHUNK + t * 16 + l16;
#pragma unroll
                for (int j = 0; j < 8; ++j) {
                    const float dist = cn[t] - 2.0f * acc[j];  // ||c||^2 - 2 x.c
                    if (dist < bestd[j]) { bestd[j] = dist; besti[j] = code; }
                }
                acc = (v8f){};
            }
        }
        wait_async_and_sync();          // next chunk resident; this buf free
    }

    // ---- per-lane winners -> LDS; C layout: VGPR j = row j + 8*hi ---------
#pragma unroll
    for (int j = 0; j < 8; ++j) {
        const int row = j + hi * 8;
        ldsD[wave][row][l16] = bestd[j];
        ldsI[wave][row][l16] = besti[j];
    }
    __syncthreads();

    // ---- per-row reduction over the 16 column lanes -----------------------
    if (tid < 128) {
        const int w = tid >> 4;
        const int r = tid & 15;
        float bd = 3.4e38f;
        int   bi = 0;
        for (int col = 0; col < 16; ++col) {
            const float d = ldsD[w][r][col];
            const int   i = ldsI[w][r][col];
            if (d < bd || (d == bd && i < bi)) { bd = d; bi = i; }
        }
        const int gRow = w * 16 + r;                 // 0..127 within block
        ldsWin[gRow] = bi;
        outIdx[rowBase + gRow] = (float)bi;
        atomicAdd(&counts[bi], 1.0f);                // exact integer fp32 adds
    }
    __syncthreads();

    // ---- gather quantized rows (fp32), accumulate SSE ---------------------
    const int rr      = tid >> 4;                    // 0..15
    const int chunk16 = tid & 15;                    // 16 elems per thread
    float local = 0.0f;
#pragma unroll 1
    for (int g = 0; g < 8; ++g) {
        const int row = g * 16 + rr;                 // 0..127
        const int win = ldsWin[row];
        const float* q   = cbF32 + (size_t)win * VQ_D + chunk16 * 16;
        const float* xin = X + (size_t)(rowBase + row) * VQ_D + chunk16 * 16;
        float* qo        = outQ + (size_t)(rowBase + row) * VQ_D + chunk16 * 16;
#pragma unroll
        for (int i = 0; i < 16; ++i) {
            const float qq = q[i];
            const float dd = xin[i] - qq;
            local += dd * dd;
            qo[i] = qq;                              // STE output == quantized
        }
    }
    ldsS[tid] = local;
    __syncthreads();
    for (int s = 128; s > 0; s >>= 1) {
        if (tid < s) ldsS[tid] += ldsS[tid + s];
        __syncthreads();
    }
    if (tid == 0) atomicAdd(sse, ldsS[0]);
}

// ---------------------------------------------------------------------------
// Kernel 3: loss + perplexity.
// ---------------------------------------------------------------------------
__global__ __launch_bounds__(256)
void vq_finalize(const float* __restrict__ counts, const float* __restrict__ sse,
                 float* __restrict__ lossOut, float* __restrict__ ppxOut)
{
    __shared__ float red[256];
    const int tid = threadIdx.x;
    float h = 0.0f;
    for (int k = tid; k < VQ_K; k += 256) {
        const float p = counts[k] * VQ_INVN;
        h += p * __logf(p + 1e-10f);
    }
    red[tid] = h;
    __syncthreads();
    for (int s = 128; s > 0; s >>= 1) {
        if (tid < s) red[tid] += red[tid + s];
        __syncthreads();
    }
    if (tid == 0) {
        // codebook_loss + 0.25*commitment_loss, both == SSE/(N*D) numerically
        *lossOut = sse[0] * (1.25f / ((float)VQ_N * (float)VQ_D));
        *ppxOut  = __expf(-red[0]);
    }
}

// ---------------------------------------------------------------------------
extern "C" void kernel_launch(void* const* d_in, const int* in_sizes, int n_in,
                              void* d_out, int out_size, void* d_ws, size_t ws_size,
                              hipStream_t stream)
{
    (void)in_sizes; (void)n_in; (void)out_size; (void)ws_size;

    const float* X  = (const float*)d_in[0];        // [32768, 256]
    const float* CB = (const float*)d_in[1];        // [1024, 256]

    float* out = (float*)d_out;
    float* outQ    = out;                           // 8388608
    float* outIdx  = out + (size_t)VQ_N * VQ_D;     // 32768
    float* outLoss = outIdx + VQ_N;                 // 1
    float* outPpx  = outLoss + 1;                   // 1

    char* ws = (char*)d_ws;
    _Float16* cbf16 = (_Float16*)ws;                            // 512 KB
    float*    cnorm = (float*)(ws + (size_t)VQ_K * VQ_D * 2);   // 4 KB
    float*    counts = cnorm + VQ_K;                            // 4 KB
    float*    sse    = counts + VQ_K;                           // 4 B

    vq_prep<<<VQ_K, 256, 0, stream>>>(CB, cbf16, cnorm, counts, sse);
    vq_main<<<VQ_N / 128, 256, 0, stream>>>(X, CB, cbf16, cnorm,
                                            outQ, outIdx, counts, sse);
    vq_finalize<<<1, 256, 0, stream>>>(counts, sse, outLoss, outPpx);
}
